// CrossAttention_73031623901662
// MI455X (gfx1250) — matrix operand order repaired and checked by reading, not compiled, not run on hardware
//
#include <hip/hip_runtime.h>

// ---------------------------------------------------------------------------
// CDNA5 (gfx1250) implementation of the dual-stream channel cross-attention.
// All GEMMs (1x1 convs, Gram matrices, attn@v, projections) run through
// v_wmma_f32_16x16x32_bf16 (bf16 A/B, fp32 accumulate). wave32 everywhere.
// ---------------------------------------------------------------------------

typedef __bf16 bf16;
typedef bf16  v16bf __attribute__((ext_vector_type(16)));
typedef float v8f   __attribute__((ext_vector_type(8)));

#define DIMC 192
#define C3   576
#define NHD  4
#define CHD  48
#define NB   4
#define NPIX 16384   // 128*128

__device__ __forceinline__ v8f wmma_bf16(v16bf a, v16bf b, v8f c) {
  // D = A(16x32) * B(32x16) + C, fp32 accumulate
  return __builtin_amdgcn_wmma_f32_16x16x32_bf16(false, a, false, b, (short)0, c,
                                                 false, false);
}

// --- WMMA fragment loaders (lane/slot -> K maps per CDNA5 ISA 7.12.2) ------
// A 16x32 (16-bit): lanes 0-15 row m, slots 0-7 -> K 0-7, slots 8-15 -> K 16-23
//                   lanes 16-31 row m, slots 0-7 -> K 8-15, slots 8-15 -> K 24-31

// A from row-major source (K contiguous within row of stride rs)
__device__ __forceinline__ v16bf ldA_row(const bf16* __restrict__ p, long rs,
                                         int lane, int k0) {
  const bf16* r = p + (long)(lane & 15) * rs + k0 + ((lane & 16) ? 8 : 0);
  v16bf a;
#pragma unroll
  for (int i = 0; i < 8; ++i) { a[i] = r[i]; a[i + 8] = r[i + 16]; }
  return a;
}

// A from column-major source: A[m,k] = p[k*cs + m]; zero when k >= kmax
__device__ __forceinline__ v16bf ldA_col(const bf16* __restrict__ p, long cs,
                                         int lane, int k0, int kmax) {
  int m  = lane & 15;
  int kb = k0 + ((lane & 16) ? 8 : 0);
  v16bf a;
#pragma unroll
  for (int i = 0; i < 8; ++i) {
    int k1 = kb + i, k2 = kb + 16 + i;
    a[i]     = (k1 < kmax) ? p[(long)k1 * cs + m] : (bf16)0.0f;
    a[i + 8] = (k2 < kmax) ? p[(long)k2 * cs + m] : (bf16)0.0f;
  }
  return a;
}

// B 32x16 (16-bit): lane = column n; lanes 0-15 slots 0-7 -> K 0-7, slots 8-15 -> K 16-23,
//                   lanes 16-31 slots 0-7 -> K 8-15, slots 8-15 -> K 24-31.
// Source [n][k] row-major (k contiguous, row stride rs): two 8-element runs.
__device__ __forceinline__ v16bf ldB_nk(const bf16* __restrict__ p, long rs,
                                        int lane, int k0) {
  const bf16* r = p + (long)(lane & 15) * rs + k0 + ((lane & 16) ? 8 : 0);
  v16bf b;
#pragma unroll
  for (int i = 0; i < 8; ++i) { b[i] = r[i]; b[i + 8] = r[i + 16]; }
  return b;
}

// Source [k][n] row-major: B[k,n] = p[k*rs + n]
__device__ __forceinline__ v16bf ldB_kn(const bf16* __restrict__ p, long rs,
                                        int lane, int k0) {
  int n  = lane & 15;
  int kb = k0 + ((lane & 16) ? 8 : 0);
  v16bf b;
#pragma unroll
  for (int i = 0; i < 8; ++i) { b[i] = p[(long)(kb + i) * rs + n]; b[i + 8] = p[(long)(kb + 16 + i) * rs + n]; }
  return b;
}

// ---------------------------------------------------------------------------
// Small utility kernels
// ---------------------------------------------------------------------------

__global__ void k_cvt(const float* __restrict__ s, bf16* __restrict__ d, int n) {
  int i = blockIdx.x * 256 + threadIdx.x;
  if (i < n) d[i] = (bf16)s[i];
}

// [b][C][N] fp32 -> [b][N][C] bf16 via LDS 32x32 tiles (coalesced both sides)
__global__ void k_transpose_cn(const float* __restrict__ X, bf16* __restrict__ XT) {
  __shared__ bf16 tile[32][33];
  int tx = threadIdx.x, ty = threadIdx.y;
  int n0 = blockIdx.x * 32, c0 = blockIdx.y * 32, b = blockIdx.z;
#pragma unroll
  for (int i = 0; i < 32; i += 8)
    tile[ty + i][tx] = (bf16)X[(((long)b * DIMC + c0 + ty + i) << 14) + n0 + tx];
  __syncthreads();
#pragma unroll
  for (int i = 0; i < 32; i += 8)
    XT[((long)b * NPIX + n0 + ty + i) * DIMC + c0 + tx] = tile[tx][ty + i];
}

// ---------------------------------------------------------------------------
// GEMM 1: qkv = W(576x192) @ x + b  -> Ylin bf16 [b][576][N]
// One wave computes a 48(row) x 16(col) tile; B fragment reused 3x.
// ---------------------------------------------------------------------------
__global__ void k_gemm_qkv(const bf16* __restrict__ W, const float* __restrict__ bias,
                           const bf16* __restrict__ XT, bf16* __restrict__ Y) {
  int wid  = (blockIdx.x * blockDim.x + threadIdx.x) >> 5;
  int lane = threadIdx.x & 31;
  int nt = wid / 12;              // 0..4095 column tiles over b*N
  int mb = wid % 12;              // 48-row block
  long nn0 = (long)nt * 16;
  int  ob  = mb * 48;
  const bf16* Xtile = XT + nn0 * DIMC;
  v8f a0 = {}, a1 = {}, a2 = {};
#pragma unroll
  for (int k0 = 0; k0 < DIMC; k0 += 32) {
    v16bf bf = ldB_nk(Xtile, DIMC, lane, k0);
    v16bf w0 = ldA_row(W + (long)ob * DIMC,        DIMC, lane, k0);
    v16bf w1 = ldA_row(W + (long)(ob + 16) * DIMC, DIMC, lane, k0);
    v16bf w2 = ldA_row(W + (long)(ob + 32) * DIMC, DIMC, lane, k0);
    a0 = wmma_bf16(w0, bf, a0);
    a1 = wmma_bf16(w1, bf, a1);
    a2 = wmma_bf16(w2, bf, a2);
  }
  int b = (int)(nn0 >> 14);
  int n = (int)(nn0 & 16383) + (lane & 15);
#pragma unroll
  for (int r = 0; r < 8; ++r) {
    int rr = r + ((lane >> 4) << 3);
    int o0 = ob + rr, o1 = ob + 16 + rr, o2 = ob + 32 + rr;
    Y[((long)b * C3 + o0) * NPIX + n] = (bf16)(a0[r] + bias[o0]);
    Y[((long)b * C3 + o1) * NPIX + n] = (bf16)(a1[r] + bias[o1]);
    Y[((long)b * C3 + o2) * NPIX + n] = (bf16)(a2[r] + bias[o2]);
  }
}

// ---------------------------------------------------------------------------
// Depthwise 3x3 (SAME, zero pad), bf16 in/out, fp32 math
// ---------------------------------------------------------------------------
__global__ void k_dwconv(const bf16* __restrict__ Y, const float* __restrict__ w,
                         const float* __restrict__ bias, bf16* __restrict__ O) {
  long idx = (long)blockIdx.x * 256 + threadIdx.x;
  int x = (int)(idx & 127);
  int y = (int)(idx >> 7) & 127;
  long bc = idx >> 14;
  int ch = (int)(bc % C3);
  const bf16* src = Y + (bc << 14);
  const float* wk = w + (long)ch * 9;
  float s = bias[ch];
#pragma unroll
  for (int dy = -1; dy <= 1; ++dy) {
    int yy = y + dy;
    if (yy < 0 || yy > 127) continue;
#pragma unroll
    for (int dx = -1; dx <= 1; ++dx) {
      int xx = x + dx;
      if (xx < 0 || xx > 127) continue;
      s += wk[(dy + 1) * 3 + (dx + 1)] * (float)src[yy * 128 + xx];
    }
  }
  O[idx] = (bf16)s;
}

// ---------------------------------------------------------------------------
// Inverse row norms for q and k: invn[st][qk][b][192], one block per row
// ---------------------------------------------------------------------------
__global__ void k_norms(const bf16* __restrict__ Qms, const bf16* __restrict__ Qsar,
                        float* __restrict__ invn) {
  int blk = blockIdx.x;                 // ((st*2+qk)*4 + b)*192 + c
  int c  = blk % 192;
  int b  = (blk / 192) & 3;
  int qk = (blk / 768) & 1;
  int st = blk / 1536;
  const bf16* p = (st ? Qsar : Qms) + (((long)b * C3 + qk * DIMC + c) << 14);
  float s = 0.f;
  for (int i = threadIdx.x; i < NPIX; i += 256) { float v = (float)p[i]; s += v * v; }
  __shared__ float red[256];
  red[threadIdx.x] = s;
  __syncthreads();
  for (int st2 = 128; st2 > 0; st2 >>= 1) {
    if (threadIdx.x < st2) red[threadIdx.x] += red[threadIdx.x + st2];
    __syncthreads();
  }
  if (threadIdx.x == 0) invn[blk] = 1.f / fmaxf(sqrtf(red[0]), 1e-12f);
}

// ---------------------------------------------------------------------------
// Gram matrices, split-K: S = Q K^T over n. grid = 3 attn * 16 (b,h) * 32 slices,
// 9 waves/block (one 16x16 tile each), K-chunk = 512.
// ---------------------------------------------------------------------------
__global__ void k_gram(const bf16* __restrict__ Qms, const bf16* __restrict__ Qsar,
                       float* __restrict__ Gpart) {
  int blk = blockIdx.x;
  int s  = blk & 31;
  int bh = (blk >> 5) & 15;
  int a  = blk >> 9;                // 0 ms_self, 1 sar_self, 2 cross
  int wv   = threadIdx.x >> 5;      // 0..8
  int lane = threadIdx.x & 31;
  int ti = wv / 3, tj = wv % 3;
  int b = bh >> 2, h = bh & 3;
  const bf16* Qs = (a == 0) ? Qms : Qsar;   // q stream
  const bf16* Ks = (a == 1) ? Qsar : Qms;   // k stream
  const bf16* qrow = Qs + (((long)b * C3 + h * CHD + ti * 16) << 14);
  const bf16* krow = Ks + (((long)b * C3 + DIMC + h * CHD + tj * 16) << 14);
  v8f acc = {};
  int n0 = s * 512;
  for (int kk = 0; kk < 512; kk += 32) {
    v16bf af = ldA_row(qrow, NPIX, lane, n0 + kk);
    v16bf bf = ldB_nk(krow, NPIX, lane, n0 + kk);   // B[kk,d] = K[d, n0+kk]
    acc = wmma_bf16(af, bf, acc);
  }
  float* dst = Gpart + (((long)(a * 16 + bh) * 32 + s) * 2304);
  int col = lane & 15;
#pragma unroll
  for (int r = 0; r < 8; ++r) {
    int rr = r + ((lane >> 4) << 3);
    dst[(ti * 16 + rr) * 48 + tj * 16 + col] = acc[r];
  }
}

// ---------------------------------------------------------------------------
// Reduce split-K partials, apply 1/(|q||k|) * temp, softmax over d,
// write transposed+padded A^T[64][48] bf16 (rows 48..63 zero).
// grid = 3*16*48 blocks (one per output row c), 64 threads (one per d).
// ---------------------------------------------------------------------------
__global__ void k_softmax(const float* __restrict__ Gpart, const float* __restrict__ invn,
                          const float* __restrict__ ms_temp, const float* __restrict__ sar_temp,
                          bf16* __restrict__ At) {
  int blk = blockIdx.x;
  int row = blk % 48;
  int bh  = (blk / 48) % 16;
  int a   = blk / 768;
  int t = threadIdx.x;              // d index (0..63)
  int b = bh >> 2, h = bh & 3;
  __shared__ float sc[48];
  __shared__ float ssum;
  if (t < 48) {
    const float* gp = Gpart + (long)(a * 16 + bh) * 32 * 2304 + row * 48 + t;
    float s = 0.f;
    for (int sl = 0; sl < 32; ++sl) s += gp[(long)sl * 2304];
    int qs = (a == 0) ? 0 : 1;
    int ks = (a == 1) ? 1 : 0;
    float iq = invn[((qs * 2 + 0) * NB + b) * DIMC + h * CHD + row];
    float ik = invn[((ks * 2 + 1) * NB + b) * DIMC + h * CHD + t];
    float tp = (a == 1) ? sar_temp[h] : ms_temp[h];
    sc[t] = s * iq * ik * tp;
  }
  __syncthreads();
  if (t == 0) {
    float m = -INFINITY;
    for (int i = 0; i < 48; ++i) m = fmaxf(m, sc[i]);
    float ss = 0.f;
    for (int i = 0; i < 48; ++i) { sc[i] = __expf(sc[i] - m); ss += sc[i]; }
    ssum = ss;
  }
  __syncthreads();
  bf16 o = (bf16)0.0f;
  if (t < 48) o = (bf16)(sc[t] / ssum);
  At[((long)(a * 16 + bh) * 64 + t) * 48 + row] = o;   // A^T[d][c], padded
}

// ---------------------------------------------------------------------------
// out^T = v^T (16384x48) @ A^T (48x48) -> attnT [a][b][N][192] bf16.
// Wave: 16 n-rows x 48 cols (A fragment of v^T reused for 3 col tiles).
// ---------------------------------------------------------------------------
__global__ void k_attnv(const bf16* __restrict__ Qms, const bf16* __restrict__ Qsar,
                        const bf16* __restrict__ Atp, bf16* __restrict__ attnT) {
  int wid  = (blockIdx.x * blockDim.x + threadIdx.x) >> 5;
  int lane = threadIdx.x & 31;
  int nt = wid & 1023;
  int h  = (wid >> 10) & 3;
  int b  = (wid >> 12) & 3;
  int a  = wid >> 14;
  const bf16* Q  = (a == 0) ? Qms : Qsar;     // v stream (cross uses sar_v)
  const bf16* vb = Q + (((long)b * C3 + 2 * DIMC + h * CHD) << 14);
  const bf16* At = Atp + (long)(a * 16 + b * 4 + h) * 64 * 48;
  int n0 = nt * 16;
  v8f a0 = {}, a1 = {}, a2 = {};
#pragma unroll
  for (int k0 = 0; k0 < 64; k0 += 32) {
    v16bf vf = ldA_col(vb + n0, NPIX, lane, k0, CHD);  // v^T[n,d], zero d>=48
    v16bf b0 = ldB_kn(At + 0,  48, lane, k0);
    v16bf b1 = ldB_kn(At + 16, 48, lane, k0);
    v16bf b2 = ldB_kn(At + 32, 48, lane, k0);
    a0 = wmma_bf16(vf, b0, a0);
    a1 = wmma_bf16(vf, b1, a1);
    a2 = wmma_bf16(vf, b2, a2);
  }
  int col = lane & 15;
#pragma unroll
  for (int r = 0; r < 8; ++r) {
    int n = n0 + r + ((lane >> 4) << 3);
    long base = (((long)a * NB + b) * NPIX + n) * DIMC + h * CHD;
    attnT[base + col]      = (bf16)a0[r];
    attnT[base + 16 + col] = (bf16)a1[r];
    attnT[base + 32 + col] = (bf16)a2[r];
  }
}

// catT[b][n][0..191] = ms_selfT; catT[b][n][192..383] = ms_selfT + crossT
__global__ void k_cat(const bf16* __restrict__ msT, const bf16* __restrict__ crT,
                      bf16* __restrict__ catT) {
  long i = (long)blockIdx.x * 256 + threadIdx.x;   // over 4*16384*192
  long bn = i / DIMC;
  int  c  = (int)(i % DIMC);
  bf16 m = msT[i];
  float sum = (float)m + (float)crT[i];
  catT[bn * 384 + c]        = m;
  catT[bn * 384 + DIMC + c] = (bf16)sum;
}

// ---------------------------------------------------------------------------
// Projection + residual: out = x + W(192xK) @ Bt^T + bias, fp32 output.
// ---------------------------------------------------------------------------
__global__ void k_proj(const bf16* __restrict__ W, const float* __restrict__ bias,
                       const bf16* __restrict__ Bt, int K,
                       const float* __restrict__ resid, float* __restrict__ out) {
  int wid  = (blockIdx.x * blockDim.x + threadIdx.x) >> 5;
  int lane = threadIdx.x & 31;
  int nt = wid >> 2;
  int mb = wid & 3;
  long nn0 = (long)nt * 16;
  int  ob  = mb * 48;
  const bf16* Btile = Bt + nn0 * K;
  v8f a0 = {}, a1 = {}, a2 = {};
  for (int k0 = 0; k0 < K; k0 += 32) {
    v16bf bf = ldB_nk(Btile, K, lane, k0);
    v16bf w0 = ldA_row(W + (long)ob * K,        K, lane, k0);
    v16bf w1 = ldA_row(W + (long)(ob + 16) * K, K, lane, k0);
    v16bf w2 = ldA_row(W + (long)(ob + 32) * K, K, lane, k0);
    a0 = wmma_bf16(w0, bf, a0);
    a1 = wmma_bf16(w1, bf, a1);
    a2 = wmma_bf16(w2, bf, a2);
  }
  int b = (int)(nn0 >> 14);
  int n = (int)(nn0 & 16383) + (lane & 15);
#pragma unroll
  for (int r = 0; r < 8; ++r) {
    int rr = r + ((lane >> 4) << 3);
    int o0 = ob + rr, o1 = ob + 16 + rr, o2 = ob + 32 + rr;
    long i0 = ((long)b * DIMC + o0) * NPIX + n;
    long i1 = ((long)b * DIMC + o1) * NPIX + n;
    long i2 = ((long)b * DIMC + o2) * NPIX + n;
    out[i0] = a0[r] + bias[o0] + resid[i0];
    out[i1] = a1[r] + bias[o1] + resid[i1];
    out[i2] = a2[r] + bias[o2] + resid[i2];
  }
}

// ---------------------------------------------------------------------------
extern "C" void kernel_launch(void* const* d_in, const int* in_sizes, int n_in,
                              void* d_out, int out_size, void* d_ws, size_t ws_size,
                              hipStream_t stream) {
  (void)in_sizes; (void)n_in; (void)out_size; (void)ws_size;
  const float* sar        = (const float*)d_in[0];
  const float* ms         = (const float*)d_in[1];
  const float* ms_qkv_w   = (const float*)d_in[2];
  const float* ms_qkv_b   = (const float*)d_in[3];
  const float* ms_dw_w    = (const float*)d_in[4];
  const float* ms_dw_b    = (const float*)d_in[5];
  const float* ms_temp    = (const float*)d_in[6];
  const float* sar_qkv_w  = (const float*)d_in[7];
  const float* sar_qkv_b  = (const float*)d_in[8];
  const float* sar_dw_w   = (const float*)d_in[9];
  const float* sar_dw_b   = (const float*)d_in[10];
  const float* sar_temp   = (const float*)d_in[11];
  const float* ms_proj_w  = (const float*)d_in[12];
  const float* ms_proj_b  = (const float*)d_in[13];
  const float* sar_proj_w = (const float*)d_in[14];
  const float* sar_proj_b = (const float*)d_in[15];
  float* out_sar = (float*)d_out;
  float* out_ms  = (float*)d_out + (size_t)NB * DIMC * NPIX;

  // workspace layout (all 256B-multiple sizes)
  char* w = (char*)d_ws;
  const size_t SZ_XT  = (size_t)NB * NPIX * DIMC * 2;     // 24 MB
  const size_t SZ_WQ  = (size_t)C3 * DIMC * 2;
  const size_t SZ_WPM = (size_t)DIMC * 2 * DIMC * 2;
  const size_t SZ_WPS = (size_t)DIMC * DIMC * 2;
  const size_t SZ_QKV = (size_t)NB * C3 * NPIX * 2;       // 72 MB
  const size_t SZ_INV = 2 * 2 * 4 * 192 * 4;
  const size_t SZ_GP  = (size_t)3 * 16 * 32 * 2304 * 4;   // 13.5 MB
  const size_t SZ_ATP = (size_t)3 * 16 * 64 * 48 * 2;

  bf16* XT_ms   = (bf16*)w;  w += SZ_XT;
  bf16* XT_sar  = (bf16*)w;  w += SZ_XT;
  bf16* Wq_ms   = (bf16*)w;  w += SZ_WQ;
  bf16* Wq_sar  = (bf16*)w;  w += SZ_WQ;
  bf16* Wp_ms   = (bf16*)w;  w += SZ_WPM;
  bf16* Wp_sar  = (bf16*)w;  w += SZ_WPS;
  bf16* Qkv_ms  = (bf16*)w;  w += SZ_QKV;
  bf16* Qkv_sar = (bf16*)w;  w += SZ_QKV;
  float* invn   = (float*)w; w += SZ_INV;
  float* Gpart  = (float*)w; w += SZ_GP;
  bf16* Atp     = (bf16*)w;  w += SZ_ATP;
  bf16* Region  = (bf16*)w;                                // 2*SZ_QKV bytes
  bf16* Ylin_ms  = Region;
  bf16* Ylin_sar = Region + (size_t)NB * C3 * NPIX;
  // reuse Ylin region after dwconv:
  const size_t CHUNK = (size_t)NB * NPIX * DIMC;           // one attn map
  bf16* attnT = Region;                                    // 3*CHUNK elems
  bf16* catT  = Region + 3 * CHUNK;                        // 4*N*384 elems

  // 1) weights -> bf16
  k_cvt<<<(C3 * DIMC + 255) / 256, 256, 0, stream>>>(ms_qkv_w, Wq_ms, C3 * DIMC);
  k_cvt<<<(C3 * DIMC + 255) / 256, 256, 0, stream>>>(sar_qkv_w, Wq_sar, C3 * DIMC);
  k_cvt<<<(DIMC * 384 + 255) / 256, 256, 0, stream>>>(ms_proj_w, Wp_ms, DIMC * 384);
  k_cvt<<<(DIMC * DIMC + 255) / 256, 256, 0, stream>>>(sar_proj_w, Wp_sar, DIMC * DIMC);

  // 2) transpose inputs to [b][n][c] bf16
  dim3 tg(NPIX / 32, DIMC / 32, NB), tb(32, 8);
  k_transpose_cn<<<tg, tb, 0, stream>>>(ms, XT_ms);
  k_transpose_cn<<<tg, tb, 0, stream>>>(sar, XT_sar);

  // 3) 1x1 conv GEMMs (12 mblks * 4096 ntiles = 49152 waves)
  k_gemm_qkv<<<6144, 256, 0, stream>>>(Wq_ms, ms_qkv_b, XT_ms, Ylin_ms);
  k_gemm_qkv<<<6144, 256, 0, stream>>>(Wq_sar, sar_qkv_b, XT_sar, Ylin_sar);

  // 4) depthwise 3x3
  k_dwconv<<<(int)((size_t)NB * C3 * NPIX / 256), 256, 0, stream>>>(Ylin_ms, ms_dw_w, ms_dw_b, Qkv_ms);
  k_dwconv<<<(int)((size_t)NB * C3 * NPIX / 256), 256, 0, stream>>>(Ylin_sar, sar_dw_w, sar_dw_b, Qkv_sar);

  // 5) inverse norms of q,k rows
  k_norms<<<3072, 256, 0, stream>>>(Qkv_ms, Qkv_sar, invn);

  // 6) Gram matrices (split-K over n)
  k_gram<<<1536, 288, 0, stream>>>(Qkv_ms, Qkv_sar, Gpart);

  // 7) reduce + scale + softmax -> padded A^T
  k_softmax<<<2304, 64, 0, stream>>>(Gpart, invn, ms_temp, sar_temp, Atp);

  // 8) attn @ v (transposed) -> attnT [a][b][N][192]
  k_attnv<<<6144, 256, 0, stream>>>(Qkv_ms, Qkv_sar, Atp, attnT);

  // 9) concat [ms_self ; cross+ms_self] -> catT [b][N][384]
  k_cat<<<(int)(CHUNK / 256), 256, 0, stream>>>(attnT, attnT + 2 * CHUNK, catT);

  // 10) projections + residuals (fp32 out)
  k_proj<<<2048, 256, 0, stream>>>(Wp_sar, sar_proj_b, attnT + CHUNK, DIMC, sar, out_sar);
  k_proj<<<2048, 256, 0, stream>>>(Wp_ms, ms_proj_b, catT, 2 * DIMC, ms, out_ms);
}
